// DynamicWindowSelfAttention_60017872994482
// MI455X (gfx1250) — matrix operand order repaired and checked by reading, not compiled, not run on hardware
//
#include <hip/hip_runtime.h>
#include <hip/hip_bf16.h>

// ---------------------------------------------------------------------------
// DynamicWindowSelfAttention for MI455X (gfx1250, wave32, WMMA + TDM)
//   B=4, T=2048, C=1024, H=16, D=64
// ---------------------------------------------------------------------------

#define BATCH 4
#define T_TOK 2048
#define CDIM  1024
#define NHEAD 16
#define DHEAD 64
#define N3    3072
#define MTOT  (BATCH * T_TOK)   // 8192

typedef __attribute__((ext_vector_type(16))) __bf16 v16bf;
typedef __attribute__((ext_vector_type(8)))  float  v8f;
typedef __attribute__((ext_vector_type(4)))  unsigned int v4u;
typedef __attribute__((ext_vector_type(8)))  int v8i_t;
typedef __attribute__((ext_vector_type(4)))  int v4i_t;

// ---------------------------------------------------------------------------
// Tensor Data Mover: 2D tile of bf16 (2-byte elements) global -> LDS.
//   w      : tile width in elements (row length, contiguous)
//   h      : tile height in rows
//   stride : global row stride in elements
// D# packing per CDNA5 ISA ch.8 (group0: count/lds/global/type;
// group1: data_size, tensor dims, tile dims, dim0 stride).
// 6-arg builtin form (clang-23 / therock HIP headers).
// ---------------------------------------------------------------------------
__device__ __forceinline__ void tdm_load_2d(void* lds_dst, const void* gsrc,
                                            unsigned w, unsigned h,
                                            unsigned stride) {
  const unsigned long long ga = (unsigned long long)(size_t)gsrc;
  const unsigned lds = (unsigned)(size_t)lds_dst;
  v4u g0;
  g0[0] = 1u;                                                  // count=1
  g0[1] = lds;                                                 // lds_addr
  g0[2] = (unsigned)(ga & 0xFFFFFFFFu);                        // global lo
  g0[3] = (unsigned)((ga >> 32) & 0x1FFFFFFu) | 0x80000000u;   // global hi | type=2
  v8i_t g1;
  g1[0] = 0x00010000;                                  // data_size=1 (2B), mask=0
  g1[1] = (int)((w & 0xFFFFu) << 16);                  // tensor_dim0 lo16
  g1[2] = (int)((w >> 16) | ((h & 0xFFFFu) << 16));    // dim0 hi | tensor_dim1 lo
  g1[3] = (int)((h >> 16) | (w << 16));                // dim1 hi | tile_dim0
  g1[4] = (int)(h & 0xFFFFu);                          // tile_dim1 (tile_dim2=0)
  g1[5] = (int)stride;                                 // tensor_dim0_stride lo32
  g1[6] = 0;                                           // stride hi | dim1_stride
  g1[7] = 0;
  v4i_t z4 = {};
  v8i_t z8 = {};
  __builtin_amdgcn_tensor_load_to_lds(g0, g1, z4, z4, z8, 0);
}

// ---------------------------------------------------------------------------
// Kernel 1: seg_start[b][t] = latest delimiter position <= t, else 0.
// ---------------------------------------------------------------------------
__global__ __launch_bounds__(256)
void seg_start_kernel(const long long* __restrict__ char_ids,
                      int* __restrict__ seg_start) {
  __shared__ int part[256];
  const int b = blockIdx.x;
  const long long* row = char_ids + (size_t)b * T_TOK;
  const int t0 = threadIdx.x * (T_TOK / 256);  // 8 per thread

  int last = -1;
  int local[8];
#pragma unroll
  for (int i = 0; i < 8; ++i) {
    const int t = t0 + i;
    if (row[t] == 32) last = t;
    local[i] = last;
  }
  part[threadIdx.x] = last;
  __syncthreads();
  for (int off = 1; off < 256; off <<= 1) {
    const int u = (threadIdx.x >= (unsigned)off) ? part[threadIdx.x - off] : -1;
    const int v = part[threadIdx.x];
    __syncthreads();
    part[threadIdx.x] = (u > v) ? u : v;
    __syncthreads();
  }
  const int prefix = (threadIdx.x > 0) ? part[threadIdx.x - 1] : -1;
#pragma unroll
  for (int i = 0; i < 8; ++i) {
    const int t = t0 + i;
    int s = (local[i] > prefix) ? local[i] : prefix;
    seg_start[b * T_TOK + t] = (s < 0) ? 0 : s;
  }
}

// ---------------------------------------------------------------------------
// WMMA GEMM: Cout[M,N] = A[M,K] * Bw[K,N] + bias, bf16 MACs, f32 accum.
// Block tile 128x128, 8 waves, each wave 64x32 = 4x2 C-tiles of 16x16.
// (manual LDS staging: needs f32 -> bf16 conversion in flight)
// ---------------------------------------------------------------------------
template <typename AT, typename OT>
__global__ __launch_bounds__(256)
void gemm_wmma_kernel(const AT* __restrict__ A, const float* __restrict__ Bw,
                      const float* __restrict__ bias, OT* __restrict__ Cout,
                      int M, int N, int K) {
  __shared__ __bf16 As[128][32];   // [m][k]
  __shared__ __bf16 Bt[128][32];   // [n][k] (transposed tile)

  const int tid  = threadIdx.x;
  const int lane = tid & 31;
  const int wid  = tid >> 5;
  const int m0 = blockIdx.y * 128;
  const int n0 = blockIdx.x * 128;
  const int wm = (wid & 1) * 64;
  const int wn = (wid >> 1) * 32;

  v8f acc[4][2] = {};

  const int arow = tid >> 1;
  const int acol = (tid & 1) * 16;
  const int brow = tid >> 3;
  const int bcol = (tid & 7) * 16;

  const int lhalf = lane >> 4;
  const int lmod  = lane & 15;

  for (int k0 = 0; k0 < K; k0 += 32) {
    __syncthreads();
    {
      const AT* ag = A + (size_t)(m0 + arow) * K + k0 + acol;
      __builtin_prefetch(ag + 32, 0, 1);
#pragma unroll
      for (int j = 0; j < 16; ++j) As[arow][acol + j] = (__bf16)(float)ag[j];
    }
    {
      const float* bg = Bw + (size_t)(k0 + brow) * N + n0 + bcol;
      __builtin_prefetch(bg + (size_t)32 * N, 0, 1);
#pragma unroll
      for (int j = 0; j < 16; ++j) Bt[bcol + j][brow] = (__bf16)bg[j];
    }
    __syncthreads();

    v16bf af[4];
#pragma unroll
    for (int i = 0; i < 4; ++i) {
      const __bf16* p0 = &As[wm + 16 * i + lmod][lhalf * 16];
      const __bf16* p1 = &As[wm + 16 * i + lmod][lhalf * 16 + 8];
#pragma unroll
      for (int e = 0; e < 8; ++e) { af[i][e] = p0[e]; af[i][8 + e] = p1[e]; }
    }
    v16bf bfr[2];
#pragma unroll
    for (int j = 0; j < 2; ++j) {
      const __bf16* p = &Bt[wn + 16 * j + lmod][16 * lhalf];
#pragma unroll
      for (int e = 0; e < 16; ++e) bfr[j][e] = p[e];
    }
#pragma unroll
    for (int i = 0; i < 4; ++i)
#pragma unroll
      for (int j = 0; j < 2; ++j)
        acc[i][j] = __builtin_amdgcn_wmma_f32_16x16x32_bf16(
            false, af[i], false, bfr[j], (short)0, acc[i][j], false, false);
  }

#pragma unroll
  for (int i = 0; i < 4; ++i) {
#pragma unroll
    for (int j = 0; j < 2; ++j) {
      const int n = n0 + wn + 16 * j + lmod;
      const float bv = bias[n];
#pragma unroll
      for (int r = 0; r < 8; ++r) {
        const int m = m0 + wm + 16 * i + r + 8 * lhalf;
        Cout[(size_t)m * N + n] = (OT)(acc[i][j][r] + bv);
      }
    }
  }
}

// ---------------------------------------------------------------------------
// Kernel 3: split qkv, scale Q by 1/sqrt(D), L2-normalize V rows.
// Emits Q,K as (B,H,T,D) bf16 and V transposed as (B,H,D,T) bf16 so the
// attention kernel can TDM-load plain 2D tiles for both operand shapes.
// Thread index t-fastest: all global accesses are cacheline-dense.
// ---------------------------------------------------------------------------
__global__ __launch_bounds__(256)
void reshape_norm_kernel(const __bf16* __restrict__ qkv,
                         __bf16* __restrict__ Qo, __bf16* __restrict__ Ko,
                         __bf16* __restrict__ Vto) {
  const int idx = blockIdx.x * 256 + threadIdx.x;   // h*MTOT + m
  const int h = idx / MTOT;
  const int m = idx % MTOT;
  const int b = m / T_TOK;
  const int t = m % T_TOK;
  const __bf16* row = qkv + (size_t)m * N3;
  const int bh = b * NHEAD + h;
  const size_t o  = ((size_t)bh * T_TOK + t) * DHEAD;
  const size_t ot = (size_t)bh * DHEAD * T_TOK + t;

  float ss = 0.0f;
#pragma unroll 8
  for (int d = 0; d < DHEAD; ++d) {
    const float f = (float)row[2 * CDIM + h * DHEAD + d];
    ss += f * f;
  }
  const float inv = 1.0f / (sqrtf(ss) + 1e-6f);
#pragma unroll 8
  for (int d = 0; d < DHEAD; ++d) {
    Qo[o + d] = (__bf16)(0.125f * (float)row[h * DHEAD + d]);  // 1/sqrt(64)
    Ko[o + d] = (__bf16)(float)row[CDIM + h * DHEAD + d];
    Vto[ot + (size_t)d * T_TOK] =
        (__bf16)((float)row[2 * CDIM + h * DHEAD + d] * inv);
  }
}

// ---------------------------------------------------------------------------
// Kernel 4: flash attention with per-query window [seg_start(q), q].
// Block: 64 queries, 4 waves x 16 rows. Key tiles of 32, staged by the TDM.
// Row sums of P are computed with a WMMA against an all-ones B matrix
// (every output column = row sum -> every lane holds l with no shuffles).
// ---------------------------------------------------------------------------
__global__ __launch_bounds__(128)
void flash_attn_kernel(const __bf16* __restrict__ Qg, const __bf16* __restrict__ Kg,
                       const __bf16* __restrict__ Vtg, const int* __restrict__ segstart,
                       __bf16* __restrict__ Y) {
  __shared__ __bf16 Qs[64][64];
  __shared__ __bf16 Ks[32][64];
  __shared__ __bf16 Vt[64][32];       // [d][k]
  __shared__ __bf16 Pb[4][16][32];    // per-wave P staging
  __shared__ int ss_sh[64];
  __shared__ int smin_sh;

  const int tid = threadIdx.x;
  const int lane = tid & 31, wid = tid >> 5;
  const int bh = blockIdx.x;          // b*H + h
  const int b = bh >> 4;
  const int h = bh & 15;
  const int q0 = blockIdx.y * 64;
  const size_t base  = (size_t)bh * T_TOK * DHEAD;   // Q/K (B,H,T,D)
  const size_t vbase = (size_t)bh * DHEAD * T_TOK;   // Vt  (B,H,D,T)

  // stage Q tile (64x64) via Tensor Data Mover (wave 0, uniform branch)
  if (wid == 0) {
    tdm_load_2d(&Qs[0][0], Qg + base + (size_t)q0 * DHEAD, 64, 64, 64);
    __builtin_amdgcn_s_wait_tensorcnt(0);
  }
  if (tid < 64) ss_sh[tid] = segstart[b * T_TOK + q0 + tid];
  __syncthreads();
  if (tid == 0) {
    int mn = ss_sh[0];
    for (int i = 1; i < 64; ++i) mn = min(mn, ss_sh[i]);
    smin_sh = mn;
  }
  __syncthreads();
  const int k_lo = smin_sh & ~31;
  const int k_hi = q0 + 64;

  const int lhalf = lane >> 4, lmod = lane & 15;

  // Q A-fragments (two d-halves), rows q0 + wid*16 + lane%16
  v16bf qa[2];
  {
    const int qr = wid * 16 + lmod;
#pragma unroll
    for (int hf = 0; hf < 2; ++hf) {
      const __bf16* p0 = &Qs[qr][hf * 32 + lhalf * 16];
      const __bf16* p1 = &Qs[qr][hf * 32 + lhalf * 16 + 8];
#pragma unroll
      for (int e = 0; e < 8; ++e) { qa[hf][e] = p0[e]; qa[hf][8 + e] = p1[e]; }
    }
  }

  v16bf ones;
#pragma unroll
  for (int e = 0; e < 16; ++e) ones[e] = (__bf16)1.0f;

  float mrow[8];
  int qidx[8], qss[8];
#pragma unroll
  for (int r = 0; r < 8; ++r) {
    mrow[r] = -3.0e38f;
    const int rr = wid * 16 + r + 8 * lhalf;
    qidx[r] = q0 + rr;
    qss[r] = ss_sh[rr];
  }
  v8f oacc[4] = {};
  v8f lacc = {};

  for (int k0 = k_lo; k0 < k_hi; k0 += 32) {
    __syncthreads();   // previous tile fully consumed before TDM overwrites LDS
    if (wid == 0) {
      tdm_load_2d(&Ks[0][0], Kg + base + (size_t)k0 * DHEAD, 64, 32, 64);
      tdm_load_2d(&Vt[0][0], Vtg + vbase + k0, 32, 64, T_TOK);
      __builtin_amdgcn_s_wait_tensorcnt(0);
    }
    __syncthreads();

    // S0: keys k0..k0+15 ; S1: k0+16..k0+31 ; contraction over d = 2 WMMAs
    v8f s0 = {}, s1 = {};
#pragma unroll
    for (int hf = 0; hf < 2; ++hf) {
      v16bf kb0, kb1;
      const __bf16* p0 = &Ks[lmod][hf * 32 + 16 * lhalf];
      const __bf16* p1 = &Ks[16 + lmod][hf * 32 + 16 * lhalf];
#pragma unroll
      for (int e = 0; e < 16; ++e) { kb0[e] = p0[e]; kb1[e] = p1[e]; }
      s0 = __builtin_amdgcn_wmma_f32_16x16x32_bf16(false, qa[hf], false, kb0,
                                                   (short)0, s0, false, false);
      s1 = __builtin_amdgcn_wmma_f32_16x16x32_bf16(false, qa[hf], false, kb1,
                                                   (short)0, s1, false, false);
    }

    const int kc0 = k0 + lmod, kc1 = k0 + 16 + lmod;
#pragma unroll
    for (int r = 0; r < 8; ++r) {
      const bool v0 = (kc0 <= qidx[r]) && (kc0 >= qss[r]);
      const bool v1 = (kc1 <= qidx[r]) && (kc1 >= qss[r]);
      float mt = fmaxf(v0 ? (float)s0[r] : -3.0e38f,
                       v1 ? (float)s1[r] : -3.0e38f);
#pragma unroll
      for (int off = 1; off < 16; off <<= 1)
        mt = fmaxf(mt, __shfl_xor(mt, off, 32));
      const float mn = fmaxf(mrow[r], mt);
      const float scale = __expf(mrow[r] - mn);
      const float p0v = v0 ? __expf((float)s0[r] - mn) : 0.0f;
      const float p1v = v1 ? __expf((float)s1[r] - mn) : 0.0f;
      mrow[r] = mn;
#pragma unroll
      for (int n = 0; n < 4; ++n) oacc[n][r] = oacc[n][r] * scale;
      lacc[r] = lacc[r] * scale;
      const int row = r + 8 * lhalf;
      Pb[wid][row][lmod] = (__bf16)p0v;
      Pb[wid][row][16 + lmod] = (__bf16)p1v;
    }

    // C-layout -> A-layout re-fragmentation of P via wave-private LDS
    v16bf pa;
    {
      const __bf16* p0 = &Pb[wid][lmod][lhalf * 16];
      const __bf16* p1 = &Pb[wid][lmod][lhalf * 16 + 8];
#pragma unroll
      for (int e = 0; e < 8; ++e) { pa[e] = p0[e]; pa[8 + e] = p1[e]; }
    }
    // row sums: l += P @ ones (every lane receives its row's sum)
    lacc = __builtin_amdgcn_wmma_f32_16x16x32_bf16(false, pa, false, ones,
                                                   (short)0, lacc, false, false);
#pragma unroll
    for (int n = 0; n < 4; ++n) {
      v16bf vb;
      const __bf16* p = &Vt[n * 16 + lmod][16 * lhalf];
#pragma unroll
      for (int e = 0; e < 16; ++e) vb[e] = p[e];
      oacc[n] = __builtin_amdgcn_wmma_f32_16x16x32_bf16(false, pa, false, vb,
                                                        (short)0, oacc[n], false, false);
    }
  }

  // finalize: y laid out (B,T,C) for the output projection
#pragma unroll
  for (int r = 0; r < 8; ++r) {
    const float lr = lacc[r];
    const float inv = (lr > 0.0f) ? 1.0f / lr : 0.0f;
    const int t = qidx[r];
#pragma unroll
    for (int n = 0; n < 4; ++n) {
      const int col = h * DHEAD + n * 16 + lmod;
      Y[((size_t)b * T_TOK + t) * CDIM + col] = (__bf16)(oacc[n][r] * inv);
    }
  }
}

// ---------------------------------------------------------------------------
// Host-side launcher
// ---------------------------------------------------------------------------
extern "C" void kernel_launch(void* const* d_in, const int* in_sizes, int n_in,
                              void* d_out, int out_size, void* d_ws, size_t ws_size,
                              hipStream_t stream) {
  const float*     x        = (const float*)d_in[0];
  const long long* char_ids = (const long long*)d_in[1];
  const float*     Wqkv     = (const float*)d_in[2];
  const float*     bqkv     = (const float*)d_in[3];
  const float*     Wo       = (const float*)d_in[4];
  const float*     bo       = (const float*)d_in[5];
  float*           out      = (float*)d_out;

  char* ws = (char*)d_ws;
  int*    segs = (int*)ws;                                   // 32 KB
  __bf16* qkv  = (__bf16*)(ws + (1 << 16));                  // 8192x3072 bf16 = 48 MB
  char*   p2   = ws + (1 << 16) + (size_t)MTOT * N3 * 2;
  __bf16* Qb  = (__bf16*)p2;                                 // (B,H,T,D) 16 MB
  __bf16* Kb  = Qb + (size_t)BATCH * NHEAD * T_TOK * DHEAD;  // (B,H,T,D) 16 MB
  __bf16* Vtb = Kb + (size_t)BATCH * NHEAD * T_TOK * DHEAD;  // (B,H,D,T) 16 MB
  __bf16* Yb  = qkv;  // reuse qkv region for attention output (B,T,C) bf16

  // 1) segment windows
  seg_start_kernel<<<BATCH, 256, 0, stream>>>(char_ids, segs);

  // 2) QKV projection: [8192,1024] x [1024,3072]
  dim3 g1(N3 / 128, MTOT / 128);
  gemm_wmma_kernel<float, __bf16><<<g1, 256, 0, stream>>>(
      x, Wqkv, bqkv, qkv, MTOT, N3, CDIM);

  // 3) split / scale / normalize-V (+ V transpose for TDM tiles)
  reshape_norm_kernel<<<(MTOT * NHEAD) / 256, 256, 0, stream>>>(qkv, Qb, Kb, Vtb);

  // 4) windowed flash attention
  dim3 g2(BATCH * NHEAD, T_TOK / 64);
  flash_attn_kernel<<<g2, 128, 0, stream>>>(Qb, Kb, Vtb, segs, Yb);

  // 5) output projection: [8192,1024] x [1024,1024] -> fp32
  dim3 g3(CDIM / 128, MTOT / 128);
  gemm_wmma_kernel<__bf16, float><<<g3, 256, 0, stream>>>(
      Yb, Wo, bo, out, MTOT, CDIM, CDIM);
}
